// PointNet2_19567871001076
// MI455X (gfx1250) — compile-verified, matrix operand order checked by reading, not run on hardware
//
#include <hip/hip_runtime.h>
#include <hip/hip_bf16.h>

typedef _Float16 h16;
typedef __attribute__((ext_vector_type(16))) _Float16 v16h;
typedef __attribute__((ext_vector_type(8)))  _Float16 v8h;
typedef __attribute__((ext_vector_type(8)))  float    v8f;

// ---------------------------------------------------------------------------
// WMMA helpers (CDNA5 gfx1250, wave32). D = A(16x32 f16) * B(32x16 f16) + C(f32)
// Layouts per CDNA5 ISA 7.12.2:
//   A frag: lane l holds row m=l&15; K = kb..kb+7 and kb+16..kb+23, kb=(l>>4)*8
//   B frag: lane l holds col n=l&15; same K split
//   C/D   : VGPR r holds (m = r + 8*(l>>4), n = l&15)
// ---------------------------------------------------------------------------
__device__ __forceinline__ v16h make_frag(v8h lo, v8h hi) {
  v16h r;
#pragma unroll
  for (int i = 0; i < 8; ++i) { r[i] = lo[i]; r[i + 8] = hi[i]; }
  return r;
}

__device__ __forceinline__ v16h load_frag_a(const h16* X, int ld, int m0, int k0) {
  const int lane = threadIdx.x & 31;
  const int m  = m0 + (lane & 15);
  const int kb = k0 + ((lane >> 4) << 3);
  const h16* p = X + (size_t)m * ld + kb;
  return make_frag(*(const v8h*)p, *(const v8h*)(p + 16));
}

__device__ __forceinline__ v16h load_frag_b(const h16* W, int ld, int n0, int k0) {
  const int lane = threadIdx.x & 31;
  const int n  = n0 + (lane & 15);
  const int kb = k0 + ((lane >> 4) << 3);
  const h16* p = W + (size_t)n * ld + kb;
  return make_frag(*(const v8h*)p, *(const v8h*)(p + 16));
}

__device__ __forceinline__ v8f wmma_f16(v16h a, v16h b, v8f c) {
  return __builtin_amdgcn_wmma_f32_16x16x32_f16(false, a, false, b, (short)0, c,
                                                false, false);
}

// y = relu(acc*s + b) -> f16 store at [m0.., n0..] with row stride ld
__device__ __forceinline__ void store_d_relu_f16(h16* Y, int ld, int m0, int n0,
                                                 v8f acc, const float* s,
                                                 const float* bia) {
  const int lane = threadIdx.x & 31;
  const int n = n0 + (lane & 15);
  const int mb = m0 + ((lane >> 4) << 3);
  const float sc = s[n], bi = bia[n];
#pragma unroll
  for (int r = 0; r < 8; ++r) {
    float v = acc[r] * sc + bi;
    v = v > 0.f ? v : 0.f;
    Y[(size_t)(mb + r) * ld + n] = (h16)v;
  }
}

// max over the 8 row-values this lane owns, post relu(s,b)
__device__ __forceinline__ float d_relu_rowmax(v8f acc, float sc, float bi) {
  float mx = 0.f;
#pragma unroll
  for (int r = 0; r < 8; ++r) {
    float v = acc[r] * sc + bi;
    v = v > 0.f ? v : 0.f;
    mx = fmaxf(mx, v);
  }
  return mx;
}

// ---------------------------------------------------------------------------
// Weight packing: f32 [O,K] -> f16 [O,Kp] zero-padded (Kp = ceil32(K))
// ---------------------------------------------------------------------------
__global__ void pack_w_kernel(const float* __restrict__ W, h16* __restrict__ Wp,
                              int O, int K, int Kp) {
  int i = blockIdx.x * 256 + threadIdx.x;
  if (i >= O * Kp) return;
  int o = i / Kp, k = i - o * Kp;
  Wp[i] = (k < K) ? (h16)W[(size_t)o * K + k] : (h16)0.f;
}

// ---------------------------------------------------------------------------
// Farthest point sampling: one block per batch, dmin resident in LDS
// ---------------------------------------------------------------------------
__global__ __launch_bounds__(256) void fps_kernel(const float* __restrict__ xyz,
                                                  int* __restrict__ fidx, int N,
                                                  int S) {
  __shared__ float dmin[8192];
  __shared__ float rv[256];
  __shared__ int ri[256];
  __shared__ int sfar;
  const int b = blockIdx.x, tid = threadIdx.x;
  const float* px = xyz + (size_t)b * N * 3;
  for (int i = tid; i < N; i += 256) dmin[i] = 1e10f;
  if (tid == 0) sfar = 0;
  __syncthreads();
  for (int p = 0; p < S; ++p) {
    const int far = sfar;
    if (tid == 0) fidx[b * S + p] = far;
    const float cx = px[far * 3], cy = px[far * 3 + 1], cz = px[far * 3 + 2];
    float bv = -1.f;
    int bi = 0;
    for (int i = tid; i < N; i += 256) {
      float dx = px[i * 3] - cx, dy = px[i * 3 + 1] - cy, dz = px[i * 3 + 2] - cz;
      float dm = fminf(dmin[i], dx * dx + dy * dy + dz * dz);
      dmin[i] = dm;
      if (dm > bv) { bv = dm; bi = i; }  // i ascending -> first-max tie-break
    }
    rv[tid] = bv; ri[tid] = bi;
    __syncthreads();
    for (int off = 128; off > 0; off >>= 1) {
      if (tid < off) {
        float ov = rv[tid + off]; int oi = ri[tid + off];
        if (ov > rv[tid] || (ov == rv[tid] && oi < ri[tid])) { rv[tid] = ov; ri[tid] = oi; }
      }
      __syncthreads();
    }
    if (tid == 0) sfar = ri[0];
    __syncthreads();
  }
}

__global__ void gather_xyz_kernel(const float* __restrict__ xyz,
                                  const int* __restrict__ fidx,
                                  float* __restrict__ out, int N, int S,
                                  int total) {
  int t = blockIdx.x * 256 + threadIdx.x;
  if (t >= total) return;
  int b = t / S;
  int pi = fidx[t];
  const float* p = xyz + ((size_t)b * N + pi) * 3;
  out[t * 3 + 0] = p[0]; out[t * 3 + 1] = p[1]; out[t * 3 + 2] = p[2];
}

// ball query: ascending-index first-NS within radius, pad with first (N-1 if none)
__global__ void ball_query_kernel(const float* __restrict__ xyz,
                                  const float* __restrict__ nxyz,
                                  int* __restrict__ idx, int N, int S, int NS,
                                  float r2, int total) {
  int t = blockIdx.x * 256 + threadIdx.x;
  if (t >= total) return;
  int b = t / S;
  const float qx = nxyz[t * 3], qy = nxyz[t * 3 + 1], qz = nxyz[t * 3 + 2];
  const float* px = xyz + (size_t)b * N * 3;
  int* o = idx + (size_t)t * NS;
  int cnt = 0, first = N - 1;
  for (int i = 0; i < N && cnt < NS; ++i) {
    float dx = px[i * 3] - qx, dy = px[i * 3 + 1] - qy, dz = px[i * 3 + 2] - qz;
    if (dx * dx + dy * dy + dz * dz < r2) {
      if (cnt == 0) first = i;
      o[cnt++] = i;
    }
  }
  for (; cnt < NS; ++cnt) o[cnt] = first;
}

// ---------------------------------------------------------------------------
// Fused SA module: one block per (group s, batch b): gather+center -> LDS,
// 3 WMMA MLP layers LDS-resident, fused max-pool over the NS points.
// ---------------------------------------------------------------------------
template <int M, int K, int NOUT>
__device__ __forceinline__ void gemm_lds(const h16* X, const h16* W,
                                         const float* s, const float* bia,
                                         h16* Y) {
  const int wave = threadIdx.x >> 5;
  const int MT = M / 16, NT = NOUT / 16;
  for (int t = wave; t < MT * NT; t += 8) {
    const int mt = t % MT, nt = t / MT;
    v8f acc = {};
#pragma unroll
    for (int k0 = 0; k0 < K; k0 += 32) {
      v16h a = load_frag_a(X, K, mt * 16, k0);
      v16h bf = load_frag_b(W, K, nt * 16, k0);
      acc = wmma_f16(a, bf, acc);
    }
    store_d_relu_f16(Y, NOUT, mt * 16, nt * 16, acc, s, bia);
  }
}

template <int M, int K, int NOUT>
__device__ __forceinline__ void gemm_maxpool(const h16* X, const h16* W,
                                             const float* s, const float* bia,
                                             h16* out) {
  const int wave = threadIdx.x >> 5, lane = threadIdx.x & 31;
  const int NT = NOUT / 16;
  for (int nt = wave; nt < NT; nt += 8) {
    const int n = nt * 16 + (lane & 15);
    const float sc = s[n], bi = bia[n];
    float mx = 0.f;
#pragma unroll
    for (int mt = 0; mt < M / 16; ++mt) {
      v8f acc = {};
#pragma unroll
      for (int k0 = 0; k0 < K; k0 += 32) {
        v16h a = load_frag_a(X, K, mt * 16, k0);
        v16h bf = load_frag_b(W, K, nt * 16, k0);
        acc = wmma_f16(a, bf, acc);
      }
      mx = fmaxf(mx, d_relu_rowmax(acc, sc, bi));
    }
    mx = fmaxf(mx, __shfl_xor(mx, 16, 32));
    if (lane < 16) out[n] = (h16)mx;
  }
}

template <int NS, int CIN, int KP0, int C1, int C2, int C3>
__global__ __launch_bounds__(256) void sa_fused_kernel(
    const float* __restrict__ xyz, const h16* __restrict__ feats,
    const float* __restrict__ nxyz, const int* __restrict__ idx,
    const h16* w1, const float* s1, const float* b1,
    const h16* w2, const float* s2, const float* b2,
    const h16* w3, const float* s3, const float* b3,
    h16* __restrict__ out, int N, int S) {
  __shared__ h16 X0[NS * KP0];
  __shared__ h16 X1[NS * C1];
  __shared__ h16 X2[NS * C2];
  const int sg = blockIdx.x, b = blockIdx.y, tid = threadIdx.x;
  const int g = b * S + sg;
  const float cx = nxyz[g * 3], cy = nxyz[g * 3 + 1], cz = nxyz[g * 3 + 2];
  for (int i = tid; i < NS * KP0; i += 256) X0[i] = (h16)0.f;
  __syncthreads();
  const int CF = CIN - 3;
  for (int t = tid; t < NS * CIN; t += 256) {
    const int j = t / CIN, c = t - j * CIN;
    const int pi = idx[(size_t)g * NS + j];
    float v;
    if (c == 0)      v = xyz[((size_t)b * N + pi) * 3 + 0] - cx;
    else if (c == 1) v = xyz[((size_t)b * N + pi) * 3 + 1] - cy;
    else if (c == 2) v = xyz[((size_t)b * N + pi) * 3 + 2] - cz;
    else             v = (float)feats[((size_t)b * N + pi) * CF + (c - 3)];
    X0[j * KP0 + c] = (h16)v;
  }
  __syncthreads();
  gemm_lds<NS, KP0, C1>(X0, w1, s1, b1, X1);
  __syncthreads();
  gemm_lds<NS, C1, C2>(X1, w2, s2, b2, X2);
  __syncthreads();
  gemm_maxpool<NS, C2, C3>(X2, w3, s3, b3, out + (size_t)g * C3);
}

// ---------------------------------------------------------------------------
// Generic tiled GEMM: Y[M,O] = relu(X[M,K] * W[O,K]^T * s + b)
// block = 64x64 tile, 8 waves, 2 accumulators per wave; X tile staged in LDS.
// mode 0: f16 Y[M,O].  mode 1: f32 transposed out[B, O, Npts] (final FC).
// ---------------------------------------------------------------------------
__global__ __launch_bounds__(256) void gemm_relu_kernel(
    const h16* __restrict__ X, const h16* __restrict__ W,
    const float* __restrict__ s, const float* __restrict__ bia,
    h16* __restrict__ Y, float* __restrict__ outT, int M, int K, int O,
    int Npts, int mode) {
  __shared__ h16 Xs[64 * 32];
  const int m0 = blockIdx.x * 64;
  const int n0 = blockIdx.y * 64;
  const int wave = threadIdx.x >> 5, lane = threadIdx.x & 31;
  const int mt = wave >> 1;
  const int ntp = (wave & 1) * 2;
  v8f acc0 = {}, acc1 = {};
  const int lr = threadIdx.x >> 2, lc = (threadIdx.x & 3) * 8;
  for (int k0 = 0; k0 < K; k0 += 32) {
    *(v8h*)(Xs + lr * 32 + lc) = *(const v8h*)(X + (size_t)(m0 + lr) * K + k0 + lc);
    if (k0 + 32 < K)
      __builtin_prefetch(X + (size_t)(m0 + lr) * K + k0 + 32 + lc, 0, 1);
    __syncthreads();
    v16h a  = load_frag_a(Xs, 32, mt * 16, 0);
    v16h b0 = load_frag_b(W, K, n0 + ntp * 16, k0);
    v16h b1 = load_frag_b(W, K, n0 + (ntp + 1) * 16, k0);
    acc0 = wmma_f16(a, b0, acc0);
    acc1 = wmma_f16(a, b1, acc1);
    __syncthreads();
  }
  const int mt0 = m0 + mt * 16;
  if (mode == 0) {
    store_d_relu_f16(Y, O, mt0, n0 + ntp * 16, acc0, s, bia);
    store_d_relu_f16(Y, O, mt0, n0 + (ntp + 1) * 16, acc1, s, bia);
  } else {
    const int mb = mt0 + ((lane >> 4) << 3);
#pragma unroll
    for (int half = 0; half < 2; ++half) {
      const v8f acc = half ? acc1 : acc0;
      const int n = n0 + (ntp + half) * 16 + (lane & 15);
      const float sc = s[n], bi = bia[n];
#pragma unroll
      for (int r = 0; r < 8; ++r) {
        const int m = mb + r;
        const int bb = m / Npts, pt = m - bb * Npts;
        float v = acc[r] * sc + bi;
        v = v > 0.f ? v : 0.f;
        outT[((size_t)bb * O + n) * Npts + pt] = v;
      }
    }
  }
}

// max over Mp rows per batch: out[b,c] = max_m Y[b*Mp+m, c]
__global__ void maxpool_kernel(const h16* __restrict__ Y, h16* __restrict__ out,
                               int Mp, int C, int total) {
  int t = blockIdx.x * 256 + threadIdx.x;
  if (t >= total) return;
  int b = t / C, c = t - b * C;
  float mx = 0.f;
  for (int m = 0; m < Mp; ++m)
    mx = fmaxf(mx, (float)Y[((size_t)b * Mp + m) * C + c]);
  out[t] = (h16)mx;
}

// ---------------------------------------------------------------------------
// Feature propagation helpers
// ---------------------------------------------------------------------------
__global__ void three_nn_kernel(const float* __restrict__ unk,
                                const float* __restrict__ kn,
                                int* __restrict__ nidx, float* __restrict__ nw,
                                int n, int m, int total) {
  int t = blockIdx.x * 256 + threadIdx.x;
  if (t >= total) return;
  int b = t / n;
  const float ux = unk[t * 3], uy = unk[t * 3 + 1], uz = unk[t * 3 + 2];
  const float* pk = kn + (size_t)b * m * 3;
  float d0 = 1e30f, d1 = 1e30f, d2 = 1e30f;
  int i0 = 0, i1 = 0, i2 = 0;
  for (int j = 0; j < m; ++j) {
    float dx = pk[j * 3] - ux, dy = pk[j * 3 + 1] - uy, dz = pk[j * 3 + 2] - uz;
    float d = dx * dx + dy * dy + dz * dz;
    if (d < d0) { d2 = d1; i2 = i1; d1 = d0; i1 = i0; d0 = d; i0 = j; }
    else if (d < d1) { d2 = d1; i2 = i1; d1 = d; i1 = j; }
    else if (d < d2) { d2 = d; i2 = j; }
  }
  float r0 = 1.f / (sqrtf(fmaxf(d0, 0.f)) + 1e-8f);
  float r1 = 1.f / (sqrtf(fmaxf(d1, 0.f)) + 1e-8f);
  float r2 = 1.f / (sqrtf(fmaxf(d2, 0.f)) + 1e-8f);
  float sum = r0 + r1 + r2;
  nidx[t * 3] = i0; nidx[t * 3 + 1] = i1; nidx[t * 3 + 2] = i2;
  nw[t * 3] = r0 / sum; nw[t * 3 + 1] = r1 / sum; nw[t * 3 + 2] = r2 / sum;
}

__global__ void interp_pack_kernel(const h16* __restrict__ KF,
                                   const int* __restrict__ nidx,
                                   const float* __restrict__ nw,
                                   h16* __restrict__ X, int n, int m, int C,
                                   int Kp, size_t total) {
  size_t t = (size_t)blockIdx.x * 256 + threadIdx.x;
  if (t >= total) return;
  int c = (int)(t % C);
  size_t row = t / C;
  int b = (int)(row / n);
  const int* id = nidx + row * 3;
  const float* w = nw + row * 3;
  const h16* base = KF + (size_t)b * m * C;
  float v = w[0] * (float)base[(size_t)id[0] * C + c] +
            w[1] * (float)base[(size_t)id[1] * C + c] +
            w[2] * (float)base[(size_t)id[2] * C + c];
  X[row * (size_t)Kp + c] = (h16)v;
}

__global__ void concat_feats_kernel(const h16* __restrict__ F,
                                    h16* __restrict__ X, int C, int Kp,
                                    int colOff, size_t total) {
  size_t t = (size_t)blockIdx.x * 256 + threadIdx.x;
  if (t >= total) return;
  int c = (int)(t % C);
  size_t row = t / C;
  X[row * (size_t)Kp + colOff + c] = F[t];
}

// fp1: X[b*128+i, 0:1024] = l3f[b], X[..., 1024:1280] = l2f[b,i]
__global__ void fp1_pack_kernel(const h16* __restrict__ l3f,
                                const h16* __restrict__ l2f,
                                h16* __restrict__ X, size_t total) {
  size_t t = (size_t)blockIdx.x * 256 + threadIdx.x;
  if (t >= total) return;
  int c = (int)(t % 1280);
  size_t row = t / 1280;
  int b = (int)(row >> 7);
  X[t] = (c < 1024) ? l3f[(size_t)b * 1024 + c] : l2f[row * 256 + (c - 1024)];
}

// sa3: X[b*128+i, :] = concat(l2_xyz[b,i](3), l2f[b,i](256), pad to 288)
__global__ void sa3_pack_kernel(const float* __restrict__ xyz,
                                const h16* __restrict__ f, h16* __restrict__ X,
                                size_t total) {
  size_t t = (size_t)blockIdx.x * 256 + threadIdx.x;
  if (t >= total) return;
  int c = (int)(t % 288);
  size_t row = t / 288;
  h16 v;
  if (c < 3)        v = (h16)xyz[row * 3 + c];
  else if (c < 259) v = f[row * 256 + (c - 3)];
  else              v = (h16)0.f;
  X[t] = v;
}

// ---------------------------------------------------------------------------
// Host side
// ---------------------------------------------------------------------------
extern "C" void kernel_launch(void* const* d_in, const int* in_sizes, int n_in,
                              void* d_out, int out_size, void* d_ws,
                              size_t ws_size, hipStream_t stream) {
  (void)in_sizes; (void)n_in; (void)out_size; (void)ws_size;
  const int B = 16, N0 = 8192, S1 = 512, S2 = 128, NS = 64;
  const float* x = (const float*)d_in[0];

  // jax pytree order: fc, fp1[0..1], fp2[0..1], fp3[0..2], sa1[0..2], sa2[0..2],
  // sa3[0..2]; each layer dict sorted -> (W, b, s) at 1+3j, +1, +2.
  static const int LO[17] = {128, 256, 256, 256, 128, 128, 128, 128,
                             64,  64,  128, 128, 128, 256, 256, 512, 1024};
  static const int LK[17] = {128, 1280, 256, 384, 256, 128, 128, 128,
                             3,   64,   64,  131, 128, 128, 259, 256, 512};
  int KP[17];
  for (int j = 0; j < 17; ++j) KP[j] = ((LK[j] + 31) / 32) * 32;
  auto Wf = [&](int j) { return (const float*)d_in[1 + 3 * j]; };
  auto Bf = [&](int j) { return (const float*)d_in[1 + 3 * j + 1]; };
  auto Sf = [&](int j) { return (const float*)d_in[1 + 3 * j + 2]; };

  char* ws = (char*)d_ws;
  size_t off = 0;
  auto alloc = [&](size_t bytes) -> void* {
    void* p = (void*)(ws + off);
    off += (bytes + 255) & ~(size_t)255;
    return p;
  };
  h16* wp[17];
  for (int j = 0; j < 17; ++j) wp[j] = (h16*)alloc((size_t)LO[j] * KP[j] * 2);
  int* fidx1 = (int*)alloc((size_t)B * S1 * 4);
  int* fidx2 = (int*)alloc((size_t)B * S2 * 4);
  float* nx1 = (float*)alloc((size_t)B * S1 * 3 * 4);
  float* nx2 = (float*)alloc((size_t)B * S2 * 3 * 4);
  int* idx1 = (int*)alloc((size_t)B * S1 * NS * 4);
  int* idx2 = (int*)alloc((size_t)B * S2 * NS * 4);
  h16* l1f = (h16*)alloc((size_t)B * S1 * 128 * 2);
  h16* l2f = (h16*)alloc((size_t)B * S2 * 256 * 2);
  h16* l3f = (h16*)alloc((size_t)B * 1024 * 2);
  h16* l2f2 = (h16*)alloc((size_t)B * S2 * 256 * 2);
  h16* l1f2 = (h16*)alloc((size_t)B * S1 * 128 * 2);
  int* nidx = (int*)alloc((size_t)B * N0 * 3 * 4);
  float* nw = (float*)alloc((size_t)B * N0 * 3 * 4);
  h16* bufA = (h16*)alloc((size_t)B * N0 * 128 * 2);
  h16* bufB = (h16*)alloc((size_t)B * N0 * 128 * 2);

  auto cdiv = [](size_t a, size_t b) { return (unsigned)((a + b - 1) / b); };

  // pack weights to padded f16
  for (int j = 0; j < 17; ++j) {
    int tot = LO[j] * KP[j];
    pack_w_kernel<<<cdiv(tot, 256), 256, 0, stream>>>(Wf(j), wp[j], LO[j], LK[j], KP[j]);
  }

  // generic GEMM launcher
  auto gemm = [&](const h16* X, int j, h16* Y, int M, int mode, float* outT,
                  int Npts) {
    dim3 g(M / 64, LO[j] / 64);
    gemm_relu_kernel<<<g, 256, 0, stream>>>(X, wp[j], Sf(j), Bf(j), Y, outT, M,
                                            KP[j], LO[j], Npts, mode);
  };

  // ---- SA1 ----
  fps_kernel<<<B, 256, 0, stream>>>(x, fidx1, N0, S1);
  gather_xyz_kernel<<<cdiv(B * S1, 256), 256, 0, stream>>>(x, fidx1, nx1, N0, S1, B * S1);
  ball_query_kernel<<<cdiv(B * S1, 256), 256, 0, stream>>>(x, nx1, idx1, N0, S1, NS, 0.2f * 0.2f, B * S1);
  sa_fused_kernel<64, 3, 32, 64, 64, 128><<<dim3(S1, B), 256, 0, stream>>>(
      x, (const h16*)nullptr, nx1, idx1, wp[8], Sf(8), Bf(8), wp[9], Sf(9),
      Bf(9), wp[10], Sf(10), Bf(10), l1f, N0, S1);

  // ---- SA2 ----
  fps_kernel<<<B, 256, 0, stream>>>(nx1, fidx2, S1, S2);
  gather_xyz_kernel<<<cdiv(B * S2, 256), 256, 0, stream>>>(nx1, fidx2, nx2, S1, S2, B * S2);
  ball_query_kernel<<<cdiv(B * S2, 256), 256, 0, stream>>>(nx1, nx2, idx2, S1, S2, NS, 0.4f * 0.4f, B * S2);
  sa_fused_kernel<64, 131, 160, 128, 128, 256><<<dim3(S2, B), 256, 0, stream>>>(
      nx1, l1f, nx2, idx2, wp[11], Sf(11), Bf(11), wp[12], Sf(12), Bf(12),
      wp[13], Sf(13), Bf(13), l2f, S1, S2);

  // ---- SA3 (global): 2048 x (288 -> 256 -> 512 -> 1024) then maxpool ----
  {
    size_t tot = (size_t)B * S2 * 288;
    sa3_pack_kernel<<<cdiv(tot, 256), 256, 0, stream>>>(nx2, l2f, bufA, tot);
    gemm(bufA, 14, bufB, B * S2, 0, nullptr, 0);
    gemm(bufB, 15, bufA, B * S2, 0, nullptr, 0);
    gemm(bufA, 16, bufB, B * S2, 0, nullptr, 0);
    maxpool_kernel<<<cdiv(B * 1024, 256), 256, 0, stream>>>(bufB, l3f, S2, 1024, B * 1024);
  }

  // ---- FP1: concat(broadcast l3f, l2f) -> 1280 -> 256 -> 256 ----
  {
    size_t tot = (size_t)B * S2 * 1280;
    fp1_pack_kernel<<<cdiv(tot, 256), 256, 0, stream>>>(l3f, l2f, bufA, tot);
    gemm(bufA, 1, bufB, B * S2, 0, nullptr, 0);
    gemm(bufB, 2, l2f2, B * S2, 0, nullptr, 0);
  }

  // ---- FP2: 3-NN interp l2f2 (256) + l1f (128) -> 384 -> 256 -> 128 ----
  {
    three_nn_kernel<<<cdiv(B * S1, 256), 256, 0, stream>>>(nx1, nx2, nidx, nw, S1, S2, B * S1);
    size_t ti = (size_t)B * S1 * 256;
    interp_pack_kernel<<<cdiv(ti, 256), 256, 0, stream>>>(l2f2, nidx, nw, bufA, S1, S2, 256, 384, ti);
    size_t tc = (size_t)B * S1 * 128;
    concat_feats_kernel<<<cdiv(tc, 256), 256, 0, stream>>>(l1f, bufA, 128, 384, 256, tc);
    gemm(bufA, 3, bufB, B * S1, 0, nullptr, 0);
    gemm(bufB, 4, l1f2, B * S1, 0, nullptr, 0);
  }

  // ---- FP3: 3-NN interp l1f2 (128) -> 128 -> 128 -> 128 ----
  {
    three_nn_kernel<<<cdiv(B * N0, 256), 256, 0, stream>>>(x, nx1, nidx, nw, N0, S1, B * N0);
    size_t ti = (size_t)B * N0 * 128;
    interp_pack_kernel<<<cdiv(ti, 256), 256, 0, stream>>>(l1f2, nidx, nw, bufA, N0, S1, 128, 128, ti);
    gemm(bufA, 5, bufB, B * N0, 0, nullptr, 0);
    gemm(bufB, 6, bufA, B * N0, 0, nullptr, 0);
    gemm(bufA, 7, bufB, B * N0, 0, nullptr, 0);
  }

  // ---- FC: 128 -> 128, relu, transposed f32 output [B, 128, 8192] ----
  gemm(bufB, 0, bufA, B * N0, 1, (float*)d_out, N0);
}
